// StatsMode_18940805775889
// MI455X (gfx1250) — compile-verified, matrix operand order checked by reading, not run on hardware
//
#include <hip/hip_runtime.h>

typedef __attribute__((ext_vector_type(16))) _Float16 v16h;
typedef __attribute__((ext_vector_type(8)))  float    v8f;
typedef __attribute__((ext_vector_type(4)))  float    v4f;

namespace {
constexpr int kN = 4096;                                    // columns (harness-fixed)
constexpr int kRowsPerWave  = 16;                           // one WMMA row-tile per wave
constexpr int kWavesPerBlock = 8;                           // 256 threads = 8 wave32
constexpr int kThreads = kWavesPerBlock * 32;
constexpr int kRowsPerBlock = kWavesPerBlock * kRowsPerWave; // 128 rows / block
}

__global__ __launch_bounds__(kThreads)
void statsmode_fill_kernel(const float* __restrict__ X,
                           float* __restrict__ out,
                           int B) {
  // Per-row sums: T = sum(X) = count1 - countInvalid ; U = sum(|X|) = count1 + countInvalid
  __shared__ float sT[kWavesPerBlock * kRowsPerWave];
  __shared__ float sU[kWavesPerBlock * kRowsPerWave];

  const int lane = threadIdx.x & 31;
  const int wave = threadIdx.x >> 5;
  const int rowBase = blockIdx.x * kRowsPerBlock + wave * kRowsPerWave;
  const bool active = rowBase < B;           // wave-uniform -> EXEC all-1s for WMMA

  const int m    = lane & 15;                // row within the 16-row tile
  const int half = lane >> 4;                // K-group select per 16-bit A layout

  v8f ct = {};                               // f32 accumulator: sum of X
  v8f cu = {};                               // f32 accumulator: sum of |X|

  if (active) {
    const float* rowPtr = X + (size_t)(rowBase + m) * kN;

    v16h bOnes;
    #pragma unroll
    for (int i = 0; i < 16; ++i) bOnes[i] = (_Float16)1.0f;

    // 16-bit A (16x32) layout: lanes 0-15 hold K = {0..7, 16..23},
    // lanes 16-31 hold K = {8..15, 24..31}  (halves 0..7 then 8..15).
    #pragma unroll 2
    for (int c0 = 0; c0 < kN; c0 += 32) {
      const float* p0 = rowPtr + c0 + half * 8;   // K = c0 + half*8 + 0..7
      const float* p1 = p0 + 16;                  // K = c0 + 16 + half*8 + 0..7
      v4f x0 = *(const v4f*)(p0 + 0);
      v4f x1 = *(const v4f*)(p0 + 4);
      v4f x2 = *(const v4f*)(p1 + 0);
      v4f x3 = *(const v4f*)(p1 + 4);

      float xv[16];
      *(v4f*)(xv + 0)  = x0;
      *(v4f*)(xv + 4)  = x1;
      *(v4f*)(xv + 8)  = x2;
      *(v4f*)(xv + 12) = x3;

      // X in {-1,0,1}: direct f16 conversion is exact; |X| via free src modifier.
      v16h at, au;
      #pragma unroll
      for (int i = 0; i < 16; ++i) {
        const float x = xv[i];
        at[i] = (_Float16)x;                    // v_cvt_f16_f32
        au[i] = (_Float16)__builtin_fabsf(x);   // v_cvt_f16_f32 |x|
      }

      // D[m,n] = sum_k A[m,k]  (B = all ones); integer sums <= 4096 exact in f32
      ct = __builtin_amdgcn_wmma_f32_16x16x32_f16(false, at, false, bOnes,
                                                  (short)0, ct, false, false);
      cu = __builtin_amdgcn_wmma_f32_16x16x32_f16(false, au, false, bOnes,
                                                  (short)0, cu, false, false);
    }

    // C/D layout: VGPR r -> (M=r, lanes 0..15) and (M=r+8, lanes 16..31);
    // every N column holds the same row-sum, so lanes 0 and 16 suffice.
    if (lane == 0) {
      #pragma unroll
      for (int r = 0; r < 8; ++r) { sT[wave*16 + r] = ct[r]; sU[wave*16 + r] = cu[r]; }
    }
    if (lane == 16) {
      #pragma unroll
      for (int r = 0; r < 8; ++r) { sT[wave*16 + 8 + r] = ct[r]; sU[wave*16 + 8 + r] = cu[r]; }
    }
  }

  __syncthreads();   // all threads reach this (no early return above)

  if (!active) return;

  #pragma unroll 1
  for (int r = 0; r < kRowsPerWave; ++r) {
    const float T = sT[wave*16 + r];
    const float U = sU[wave*16 + r];
    // count1 = (T+U)/2, count0 = N-U, countInvalid = (U-T)/2 (all exact ints)
    const float mode     = (T + 3.0f * U > 2.0f * (float)kN) ? 1.0f : 0.0f; // tie -> 0
    const bool  hasValid = (U - T) < 2.0f * (float)kN;                      // invalid < N

    const float* rp = X + (size_t)(rowBase + r) * kN;
    float*       op = out + (size_t)(rowBase + r) * (size_t)kN * 2;

    #pragma unroll 1
    for (int c = lane * 4; c < kN; c += 32 * 4) {
      v4f x = __builtin_nontemporal_load((const v4f*)(rp + c));  // last use of X

      const float o0 = hasValid ? ((x[0] != -1.0f) ? x[0] : mode) : 1.0f;
      const float o1 = hasValid ? ((x[1] != -1.0f) ? x[1] : mode) : 1.0f;
      const float o2 = hasValid ? ((x[2] != -1.0f) ? x[2] : mode) : 1.0f;
      const float o3 = hasValid ? ((x[3] != -1.0f) ? x[3] : mode) : 1.0f;

      v4f lo = { 1.0f - o0, o0, 1.0f - o1, o1 };
      v4f hi = { 1.0f - o2, o2, 1.0f - o3, o3 };
      __builtin_nontemporal_store(lo, (v4f*)(op + 2*c + 0));   // streaming 512MB
      __builtin_nontemporal_store(hi, (v4f*)(op + 2*c + 4));
    }
  }
}

extern "C" void kernel_launch(void* const* d_in, const int* in_sizes, int n_in,
                              void* d_out, int out_size, void* d_ws, size_t ws_size,
                              hipStream_t stream) {
  (void)n_in; (void)out_size; (void)d_ws; (void)ws_size;
  const float* X = (const float*)d_in[0];
  float* out = (float*)d_out;
  const int total = in_sizes[0];
  const int B = total / kN;                       // 16384 for the harness shape
  const int grid = (B + kRowsPerBlock - 1) / kRowsPerBlock;
  statsmode_fill_kernel<<<grid, kThreads, 0, stream>>>(X, out, B);
}
